// HSMNet_68444598829763
// MI455X (gfx1250) — compile-verified, matrix operand order checked by reading, not run on hardware
//
#include <hip/hip_runtime.h>
#include <hip/hip_bf16.h>

typedef __attribute__((ext_vector_type(16))) _Float16 v16h;
typedef __attribute__((ext_vector_type(8)))  float    v8f;

#define BB 4
#define CC 32
#define HH 80
#define WW 160
#define DD 24
#define NWT (WW / 16)          // 10 w-tiles per row
#define NTILES (DD * NWT)      // 240 WMMA tiles per (b,h) row
#define NTHREADS 256           // 8 wave32s

// f16 abs: folds to llvm.fabs.f16 / fabs.v2f16, which lowers to a legal
// 32-bit packed pattern. (f32 fabs is avoided entirely: the gfx1250 backend
// miscombines SLP-vectorized v2f32 fabs into an illegal scalar-i32<->v2f32
// bitcast and crashes ISel.)
static __device__ __forceinline__ _Float16 habs16(_Float16 x) {
    return x < (_Float16)0.0f ? (_Float16)(-x) : x;
}

// One block per (b,h). Stages both feature rows in LDS as f16 [w][c], builds
// the 24x160 channel-aggregated cost row via v_wmma_f32_16x16x32_f16
// (A = f16 abs-diffs, B = ones, K=32 == C), then fuses softmax-over-d +
// disparity expectation.
__global__ __launch_bounds__(NTHREADS)
void hsm_costvol_disp_kernel(const float* __restrict__ ref,
                             const float* __restrict__ tgt,
                             float* __restrict__ out) {
    __shared__ _Float16 s_ref[WW * CC];  // [w][c], 10 KB
    __shared__ _Float16 s_tgt[WW * CC];  // [w][c], 10 KB
    __shared__ float    s_cost[DD * WW]; // [d][w], 15 KB

    const int bid = blockIdx.x;
    const int b = bid / HH;
    const int h = bid % HH;
    const int tid = threadIdx.x;

    // ---- Phase 0: stage this (b,h) row of both features into LDS (f16) ----
    const size_t gbase = ((size_t)b * CC) * (size_t)(HH * WW) + (size_t)h * WW;
    for (int i = tid; i < CC * WW; i += NTHREADS) {
        const int c = i / WW;
        const int w = i % WW;              // consecutive tid -> consecutive w (coalesced)
        const size_t ga = gbase + (size_t)c * (HH * WW) + w;
        s_ref[w * CC + c] = (_Float16)ref[ga];
        s_tgt[w * CC + c] = (_Float16)tgt[ga];
    }
    __syncthreads();

    // ---- Phase 1: WMMA channel reduction -> s_cost[d][w] ----
    const int wave = tid >> 5;   // wave-uniform
    const int lane = tid & 31;
    const int l15  = lane & 15;  // A-matrix row M for this lane
    const int g    = lane >> 4;  // selects K groups {8g..8g+7} and {16+8g..23+8g}

    v16h ones;
#pragma unroll
    for (int i = 0; i < 16; ++i) ones[i] = (_Float16)1.0f;

#pragma unroll 2
    for (int t = wave; t < NTILES; t += (NTHREADS / 32)) {  // uniform trip count per wave
        const int d  = t / NWT;
        const int w0 = (t % NWT) * 16;
        const int w  = w0 + l15;     // this lane's row (w position)
        const int cb0 = 8 * g;       // K = 8g..8g+7      <-> channels cb0..cb0+7
        const int cb1 = 16 + 8 * g;  // K = 16+8g..23+8g  <-> channels cb1..cb1+7

        v16h a;
        if (w >= d) {
            const _Float16* rp = &s_ref[w * CC];
            const _Float16* tp = &s_tgt[(w - d) * CC];
#pragma unroll
            for (int i = 0; i < 8; ++i) {
                a[i]     = habs16((_Float16)(rp[cb0 + i] - tp[cb0 + i]));
                a[8 + i] = habs16((_Float16)(rp[cb1 + i] - tp[cb1 + i]));
            }
        } else {
            // invalid (w < d): reference zeros these cost entries
#pragma unroll
            for (int i = 0; i < 16; ++i) a[i] = (_Float16)0.0f;
        }

        // D = A x ones + 0 : every column n of D holds the row-sums of A.
        v8f acc = {};
        acc = __builtin_amdgcn_wmma_f32_16x16x32_f16(
            /*neg_a=*/false, a, /*neg_b=*/false, ones,
            /*c_mod=*/(short)0, acc, /*reuse_a=*/false, /*reuse_b=*/false);

        // C/D layout: lane n=0 of each half holds row-sums of rows 8g..8g+7
        // in elements 0..7 (constant indices -> no cndmask chain; the 8
        // consecutive f32 stores merge into 2x ds_store_b128).
        if (l15 == 0) {
            float* cp = &s_cost[d * WW + w0 + 8 * g];
#pragma unroll
            for (int r = 0; r < 8; ++r) cp[r] = acc[r];
        }
    }
    __syncthreads();

    // ---- Phase 2: fused softmax over d + expected disparity ----
    for (int w = tid; w < WW; w += NTHREADS) {
        float m = -1e30f;
#pragma unroll 4
        for (int d = 0; d < DD; ++d) m = fmaxf(m, s_cost[d * WW + w]);
        float sum_e = 0.0f, sum_wd = 0.0f;
#pragma unroll 4
        for (int d = 0; d < DD; ++d) {
            const float e = __expf(s_cost[d * WW + w] - m);
            sum_e  += e;
            sum_wd += e * (float)d;
        }
        out[(size_t)(b * HH + h) * WW + w] = sum_wd / sum_e;
    }
}

extern "C" void kernel_launch(void* const* d_in, const int* in_sizes, int n_in,
                              void* d_out, int out_size, void* d_ws, size_t ws_size,
                              hipStream_t stream) {
    (void)in_sizes; (void)n_in; (void)d_ws; (void)ws_size; (void)out_size;
    const float* ref = (const float*)d_in[0];   // refimg_fea   [4,32,80,160] f32
    const float* tgt = (const float*)d_in[1];   // targetimg_fea[4,32,80,160] f32
    // d_in[2] = maxdisp scalar (24), baked into the kernel constants.
    float* out = (float*)d_out;                 // pred [4,80,160] f32

    hsm_costvol_disp_kernel<<<BB * HH, NTHREADS, 0, stream>>>(ref, tgt, out);
}
